// SparseTF_4827543241279
// MI455X (gfx1250) — matrix operand order repaired and checked by reading, not compiled, not run on hardware
//
#include <hip/hip_runtime.h>

typedef __attribute__((ext_vector_type(16))) __bf16 v16bf;
typedef __attribute__((ext_vector_type(8)))  float  v8f;
typedef __attribute__((ext_vector_type(4)))  unsigned int u32x4;
typedef __attribute__((ext_vector_type(8)))  int i32x8;
typedef __attribute__((ext_vector_type(4)))  int i32x4;

#define BATCH 8192
#define DIN   4096
#define DOUT  2048

#define BM 128
#define BN 128
#define BK 32
#define LDS_STRIDE 40            // halves: 64B data + 16B pad per row
#define BUF_HALVES (BM * LDS_STRIDE)   // 5120 halves = 10240 B per buffer

// ---------------------------------------------------------------------------
// Kernel 0: bulk fp32 -> bf16 convert of the activations (one HBM pass).
// ---------------------------------------------------------------------------
union BF8 { __bf16 h[8]; uint4 u; };

__global__ __launch_bounds__(256) void convert_to_bf16(
    const float* __restrict__ in, unsigned short* __restrict__ out)
{
    const size_t idx = ((size_t)blockIdx.x * blockDim.x + threadIdx.x) * 8;
    const float4 f0 = ((const float4*)(in + idx))[0];
    const float4 f1 = ((const float4*)(in + idx))[1];
    BF8 o;
    o.h[0] = (__bf16)f0.x; o.h[1] = (__bf16)f0.y;
    o.h[2] = (__bf16)f0.z; o.h[3] = (__bf16)f0.w;
    o.h[4] = (__bf16)f1.x; o.h[5] = (__bf16)f1.y;
    o.h[6] = (__bf16)f1.z; o.h[7] = (__bf16)f1.w;
    *(uint4*)(out + idx) = o.u;
}

// ---------------------------------------------------------------------------
// Kernel 1: W't[n][k] = bf16(kernel[k][n] * map[k][n])  (mask fold + transpose)
// ---------------------------------------------------------------------------
__global__ __launch_bounds__(256) void mask_fold_transpose(
    const float* __restrict__ kern, const float* __restrict__ map,
    unsigned short* __restrict__ wt)
{
    __shared__ unsigned short tile[32][33];
    const int n0 = blockIdx.x * 32;
    const int k0 = blockIdx.y * 32;
    const int tx = threadIdx.x;   // 0..31
    const int ty = threadIdx.y;   // 0..7

#pragma unroll
    for (int i = 0; i < 4; ++i) {
        const int k = k0 + ty + 8 * i;
        const int n = n0 + tx;
        const size_t idx = (size_t)k * DOUT + n;
        tile[tx][ty + 8 * i] = __builtin_bit_cast(unsigned short, (__bf16)(kern[idx] * map[idx]));
    }
    __syncthreads();
#pragma unroll
    for (int i = 0; i < 4; ++i) {
        const int n = n0 + ty + 8 * i;
        wt[(size_t)n * DIN + k0 + tx] = tile[ty + 8 * i][tx];
    }
}

// ---------------------------------------------------------------------------
// TDM descriptors (CDNA5 ISA). Group 1 is loop-invariant
// (tile shape, pad, strides); group 0 carries the per-iteration addresses.
// ---------------------------------------------------------------------------
__device__ __forceinline__ i32x8 tdm_make_g1(unsigned tensor_rows)
{
    return (i32x8){
        (int)((1u << 16) |                    // data_size = 2 bytes
              (1u << 20) |                    // pad_enable
              (3u << 22) |                    // pad_interval: 16 DWORDs (64B row)
              (3u << 25)),                    // pad_amount: 4 DWORDs (16B pad)
        (int)((unsigned)DIN << 16),           // tensor_dim0[15:0] << 16 (=4096)
        (int)(tensor_rows << 16),             // tensor_dim1[15:0] << 16
        (int)(32u << 16),                     // tile_dim0 = 32 elements (K)
        128,                                  // tile_dim1 = 128 rows
        DIN,                                  // tensor_dim0_stride = 4096
        0, 0
    };
}

__device__ __forceinline__ void tdm_issue(unsigned lds_off,
                                          unsigned long long gaddr,
                                          i32x8 g1)
{
    const u32x4 g0 = {
        1u,                                   // count=1 valid, no gather
        lds_off,                              // LDS byte address
        (unsigned)gaddr,                      // global_addr[31:0]
        (((unsigned)(gaddr >> 32)) & 0x1FFFFFFu) | (2u << 30) // addr[56:32] | type=2
    };
    const i32x4 z4 = {0, 0, 0, 0};
    const i32x8 z8 = {0, 0, 0, 0, 0, 0, 0, 0};
    __builtin_amdgcn_tensor_load_to_lds(g0, g1, z4, z4, z8, 0);
}

// ---------------------------------------------------------------------------
// Kernel 2: C = A_bf16 @ W't + bias. 128x128 block tile, 8 waves (wave32),
// each wave 2(M) x 4(N) wmma tiles. TDM double-buffered global->LDS staging.
// ---------------------------------------------------------------------------
union FragBF { uint4 u[2]; v16bf v; };

__global__ __launch_bounds__(256) void sparse_tf_gemm(
    const unsigned short* __restrict__ A,  // [BATCH][DIN] bf16
    const unsigned short* __restrict__ Wt, // [DOUT][DIN] bf16 (masked, transposed)
    const float* __restrict__ bias,        // [DOUT]
    float* __restrict__ C)                 // [BATCH][DOUT]
{
    __shared__ unsigned short As[2 * BUF_HALVES];
    __shared__ unsigned short Bs[2 * BUF_HALVES];

    const int tid  = threadIdx.x;
    const int lane = tid & 31;
    const int wave = tid >> 5;      // 0..7
    const int wm   = wave & 3;      // 0..3 -> 32 rows
    const int wn   = wave >> 2;     // 0..1 -> 64 cols

    const int m0 = blockIdx.y * BM;
    const int n0 = blockIdx.x * BN;

    const int laneLo = lane & 15;
    const int laneHi = lane >> 4;

    v8f acc[2][4];
#pragma unroll
    for (int mt = 0; mt < 2; ++mt)
#pragma unroll
        for (int nt = 0; nt < 4; ++nt)
            acc[mt][nt] = (v8f){0.f, 0.f, 0.f, 0.f, 0.f, 0.f, 0.f, 0.f};

    const unsigned short* Abase = A  + (size_t)m0 * DIN;
    const unsigned short* Bbase = Wt + (size_t)n0 * DIN;

    const unsigned asLds = (unsigned)(unsigned long long)(const void*)&As[0];
    const unsigned bsLds = (unsigned)(unsigned long long)(const void*)&Bs[0];
    const i32x8 g1A = tdm_make_g1(BATCH);
    const i32x8 g1B = tdm_make_g1(DOUT);

    // preload tiles for k=0 into buffer 0 (wave 0 drives the TDM)
    if (wave == 0) {
        tdm_issue(asLds, (unsigned long long)(const void*)Abase, g1A);
        tdm_issue(bsLds, (unsigned long long)(const void*)Bbase, g1B);
        __builtin_amdgcn_s_wait_tensorcnt(0);
    }
    __syncthreads();

    // loop-invariant per-lane fragment base offsets (halves) within a buffer
    const int aOff = (wm * 32 + laneLo) * LDS_STRIDE + laneHi * 8;
    const int bOff = (wn * 64 + laneLo) * LDS_STRIDE + laneHi * 8;

    const int NIT = DIN / BK;   // 128
    int curOff = 0;             // halves; toggles 0 <-> BUF_HALVES
#pragma unroll 1
    for (int i = 0; i < NIT; ++i) {
        const int nxtOff = BUF_HALVES - curOff;

        // kick off DMA for the next K-tile into the other buffer
        if (i + 1 < NIT && wave == 0) {
            const int knext = (i + 1) * BK;
            tdm_issue(asLds + (unsigned)nxtOff * 2,
                      (unsigned long long)(const void*)(Abase + knext), g1A);
            tdm_issue(bsLds + (unsigned)nxtOff * 2,
                      (unsigned long long)(const void*)(Bbase + knext), g1B);
        }

        // ---- fragments per ISA 16-bit WMMA layouts ----
        // A (16x32): lanes 0-15 hold K 0..7 (V0-3) and 16..23 (V4-7); lanes 16-31: +8
        FragBF afr[2];
#pragma unroll
        for (int mt = 0; mt < 2; ++mt) {
            const int base = curOff + aOff + mt * 16 * LDS_STRIDE;
            afr[mt].u[0] = *(const uint4*)&As[base];
            afr[mt].u[1] = *(const uint4*)&As[base + 16];
        }
        // B (32x16): lanes 0-15 (col N=lane) hold K 0..7 and 16..23; lanes 16-31: +8
        FragBF bfr[4];
#pragma unroll
        for (int nt = 0; nt < 4; ++nt) {
            const int base = curOff + bOff + nt * 16 * LDS_STRIDE;
            bfr[nt].u[0] = *(const uint4*)&Bs[base];
            bfr[nt].u[1] = *(const uint4*)&Bs[base + 16];
        }

#pragma unroll
        for (int mt = 0; mt < 2; ++mt)
#pragma unroll
            for (int nt = 0; nt < 4; ++nt)
                acc[mt][nt] = __builtin_amdgcn_wmma_f32_16x16x32_bf16(
                    false, afr[mt].v, false, bfr[nt].v,
                    (short)0, acc[mt][nt], false, false);

        // next tiles must have landed before the barrier publishes them
        if (wave == 0) __builtin_amdgcn_s_wait_tensorcnt(0);
        __syncthreads();

        curOff = nxtOff;
    }

    // ---- epilogue: bias + store (C/D layout: M = r + 8*laneHi, N = laneLo) ----
#pragma unroll
    for (int nt = 0; nt < 4; ++nt) {
        const int n  = n0 + wn * 64 + nt * 16 + laneLo;
        const float bv = bias[n];
#pragma unroll
        for (int mt = 0; mt < 2; ++mt) {
            const int mbase = m0 + wm * 32 + mt * 16 + laneHi * 8;
            const v8f c = acc[mt][nt];
#pragma unroll
            for (int r = 0; r < 8; ++r)
                C[(size_t)(mbase + r) * DOUT + n] = c[r] + bv;
        }
    }
}

// ---------------------------------------------------------------------------
extern "C" void kernel_launch(void* const* d_in, const int* in_sizes, int n_in,
                              void* d_out, int out_size, void* d_ws, size_t ws_size,
                              hipStream_t stream) {
    const float* inputs = (const float*)d_in[0];  // [8192][4096]
    const float* kern   = (const float*)d_in[1];  // [4096][2048]
    const float* bias   = (const float*)d_in[2];  // [2048]
    const float* map    = (const float*)d_in[3];  // [4096][2048]
    float* out          = (float*)d_out;          // [8192][2048]

    unsigned short* a_bf = (unsigned short*)d_ws;                     // 64 MB
    unsigned short* wt   = a_bf + (size_t)BATCH * DIN;                // +16 MB

    // 0) activations fp32 -> bf16 (single HBM pass)
    const int nconv = (BATCH * DIN / 8) / 256;                        // 16384 blocks
    convert_to_bf16<<<nconv, 256, 0, stream>>>(inputs, a_bf);

    // 1) fold mask into weights, cast bf16, transpose to [DOUT][DIN]
    dim3 g1(DOUT / 32, DIN / 32), b1(32, 8);
    mask_fold_transpose<<<g1, b1, 0, stream>>>(kern, map, wt);

    // 2) WMMA GEMM + bias (TDM-staged, double-buffered)
    dim3 g2(DOUT / BN, BATCH / BM), b2(256);
    sparse_tf_gemm<<<g2, b2, 0, stream>>>(a_bf, wt, bias, out);
}
